// FullyConnectedTensorProduct_28192165331136
// MI455X (gfx1250) — matrix operand order repaired and checked by reading, not compiled, not run on hardware
//
#include <hip/hip_runtime.h>

typedef __attribute__((ext_vector_type(16))) _Float16 v16h;
typedef __attribute__((ext_vector_type(8)))  float    v8f;
typedef __attribute__((ext_vector_type(4)))  unsigned int v4u;

#define AS_STRIDE 136      // 128 K-columns + 8 pad halves (conflict-free ds_load_b128)
#define CG_STRIDE 128      // floats per path slot in workspace
#define BT_OFFSET 8192     // byte offset of f16 weight matrix in workspace

union frag16 { v16h h; v4u q[2]; };

// ---------------------------------------------------------------------------
// Setup kernel 1: exact real-basis SO(3) Clebsch-Gordan tensors (Racah formula
// + complex->real basis change with (-i)^l phase), double precision, one
// thread per path. Matches the reference's _su2_cg/_q/_so3_cg math.
// ---------------------------------------------------------------------------
__device__ double dfact(int n) { double r = 1.0; for (int i = 2; i <= n; ++i) r *= (double)i; return r; }

__device__ double su2cg(int j1, int m1, int j2, int m2, int j3, int m3) {
  if (m1 + m2 != m3) return 0.0;
  int vmin = -j1 + j2 + m3;
  if (-j1 + m1 > vmin) vmin = -j1 + m1;
  if (vmin < 0) vmin = 0;
  int vmax = j2 + j3 + m1;
  if (j3 - j1 + j2 < vmax) vmax = j3 - j1 + j2;
  if (j3 + m3 < vmax) vmax = j3 + m3;
  if (vmax < vmin) return 0.0;
  double C = sqrt((double)(2 * j3 + 1) *
                  dfact(j3 + j1 - j2) * dfact(j3 - j1 + j2) * dfact(j1 + j2 - j3) *
                  dfact(j3 + m3) * dfact(j3 - m3) /
                  (dfact(j1 + j2 + j3 + 1) * dfact(j1 - m1) * dfact(j1 + m1) *
                   dfact(j2 - m2) * dfact(j2 + m2)));
  double S = 0.0;
  for (int v = vmin; v <= vmax; ++v) {
    double t = dfact(j2 + j3 + m1 - v) * dfact(j1 - m1 + v) /
               (dfact(v) * dfact(j3 - j1 + j2 - v) * dfact(j3 + m3 - v) * dfact(v + j1 - j2 - m3));
    S += ((v + j2 + m2) & 1) ? -t : t;
  }
  return C * S;
}

struct cd { double r, i; };
__device__ cd cmul(cd a, cd b) { return {a.r * b.r - a.i * b.i, a.r * b.i + a.i * b.r}; }

__device__ void build_q(int l, double qr[5][5], double qi[5][5]) {
  for (int a = 0; a < 5; ++a) for (int b = 0; b < 5; ++b) { qr[a][b] = 0.0; qi[a][b] = 0.0; }
  const double s = 0.7071067811865476;
  for (int m = -l; m < 0; ++m) { qr[l + m][l - m] = s; qi[l + m][l + m] = -s; }
  qr[l][l] = 1.0;
  for (int m = 1; m <= l; ++m) { double g = (m & 1) ? -1.0 : 1.0; qr[l + m][l + m] = g * s; qi[l + m][l - m] = g * s; }
  int ph = l & 3;                       // multiply whole matrix by (-i)^l
  if (ph) for (int a = 0; a < 5; ++a) for (int b = 0; b < 5; ++b) {
    double r = qr[a][b], i = qi[a][b];
    if (ph == 1)      { qr[a][b] =  i; qi[a][b] = -r; }
    else if (ph == 2) { qr[a][b] = -r; qi[a][b] = -i; }
    else              { qr[a][b] = -i; qi[a][b] =  r; }
  }
}

__global__ void cg_setup(float* __restrict__ cgws) {
  const int p = threadIdx.x;
  if (p >= 11) return;
  const int L1t[11] = {0,1,2, 0,1,1,2, 0,1,2,2};
  const int L2t[11] = {0,1,2, 1,0,2,1, 2,1,0,2};
  const int LOt[11] = {0,0,0, 1,1,1,1, 2,2,2,2};
  const int l1 = L1t[p], l2 = L2t[p], lo = LOt[p];
  const int d1 = 2 * l1 + 1, d2 = 2 * l2 + 1, kp = 2 * lo + 1;
  double q1r[5][5], q1i[5][5], q2r[5][5], q2i[5][5], q3r[5][5], q3i[5][5];
  build_q(l1, q1r, q1i); build_q(l2, q2r, q2i); build_q(lo, q3r, q3i);
  for (int j = 0; j < d1; ++j)
    for (int l = 0; l < d2; ++l)
      for (int n = 0; n < kp; ++n) {
        double re = 0.0;
        for (int a = 0; a < d1; ++a)
          for (int b = 0; b < d2; ++b)
            for (int c = 0; c < kp; ++c) {
              double w = su2cg(l1, a - l1, l2, b - l2, lo, c - lo);
              if (w == 0.0) continue;
              cd t = cmul({q1r[a][j], q1i[a][j]}, {q2r[b][l], q2i[b][l]});
              t = cmul(t, {q3r[c][n], -q3i[c][n]});
              re += t.r * w;
            }
        cgws[p * CG_STRIDE + (j * d2 + l) * kp + n] = (float)re;
      }
}

// ---------------------------------------------------------------------------
// Setup kernel 2: W fp32 [p][u][v][w] -> f16 Bt [p][w][u*64+v]
// (K = uv contiguous per w column == exact per-lane B fragment layout)
// ---------------------------------------------------------------------------
__global__ void wconv(const float* __restrict__ W, _Float16* __restrict__ bt) {
  int idx = blockIdx.x * 256 + threadIdx.x;
  if (idx >= 11 * 64 * 4096) return;
  int uv = idx & 4095;
  int t  = idx >> 12;
  int w  = t & 63;
  int p  = t >> 6;
  int u = uv >> 6, v = uv & 63;
  bt[idx] = (_Float16)W[(((size_t)p * 64 + u) * 64 + v) * 64 + w];
}

// ---------------------------------------------------------------------------
// Main tensor-product kernel
// ---------------------------------------------------------------------------
__device__ constexpr int seg_off(int l) { return l == 0 ? 0 : (l == 1 ? 64 : 256); }

template <int L1, int L2, int LO>
__device__ __forceinline__ void process_path(
    int p, int zbase,
    const float* __restrict__ x1, const float* __restrict__ x2,
    const float* __restrict__ cgws, const _Float16* __restrict__ bt,
    _Float16* As, float* cg_lds, v8f* acc)
{
  constexpr int D1 = 2 * L1 + 1, D2 = 2 * L2 + 1, KP = 2 * LO + 1;
  constexpr int OFF1 = seg_off(L1), OFF2 = seg_off(L2);
  const int tid  = threadIdx.x;
  const int lane = tid & 31;
  const int wave = tid >> 5;
  const int mi = wave >> 2, ni = wave & 3;              // 2 M-tiles x 4 N-tiles
  // formation mapping: thread -> (z_local, u_local, v-quarter)
  const int pair = tid >> 2, vq = tid & 3;
  const int zl = pair >> 1, uu = pair & 1;
  const int z  = zbase + zl;
  // WMMA lane addressing (ISA 16x32 A layout / 32x16 B layout)
  const int wcol    = ni * 16 + (lane & 15);
  const size_t btrow = ((size_t)(p * 64 + wcol)) * 4096;
  const int kb_lane = (lane & 16) ? 8 : 0;
  const int Mrow    = mi * 16 + (lane & 15);

  if (tid < D1 * D2 * KP) cg_lds[tid] = cgws[p * CG_STRIDE + tid];
  __syncthreads();

  for (int c = 0; c < 32; ++c) {                        // K chunks of 128 (u = 2c, 2c+1)
    // ---- form A (tmp) chunk into LDS, f16, padded row layout, K in natural order ----
    {
      const int u = 2 * c + uu;
      float x1v[D1];
#pragma unroll
      for (int i = 0; i < D1; ++i)
        x1v[i] = x1[(size_t)z * 576 + OFF1 + u * D1 + i];
      float g[D2][KP];
#pragma unroll
      for (int j = 0; j < D2; ++j)
#pragma unroll
        for (int k = 0; k < KP; ++k) {
          float s = 0.f;
#pragma unroll
          for (int i = 0; i < D1; ++i)
            s += x1v[i] * cg_lds[(i * D2 + j) * KP + k];
          g[j][k] = s;
        }
      _Float16 hb[KP][16];
      const int vb = vq * 16;
#pragma unroll
      for (int vv = 0; vv < 16; ++vv) {
        const int v = vb + vv;
        float x2v[D2];
#pragma unroll
        for (int j = 0; j < D2; ++j)
          x2v[j] = x2[(size_t)z * 576 + OFF2 + v * D2 + j];
#pragma unroll
        for (int k = 0; k < KP; ++k) {
          float s = 0.f;
#pragma unroll
          for (int j = 0; j < D2; ++j) s += x2v[j] * g[j][k];
          hb[k][vv] = (_Float16)s;
        }
      }
#pragma unroll
      for (int k = 0; k < KP; ++k) {
        _Float16* dst = &As[(size_t)(k * 32 + zl) * AS_STRIDE + uu * 64 + vb];
        *(v4u*)(dst)     = *(const v4u*)&hb[k][0];
        *(v4u*)(dst + 8) = *(const v4u*)&hb[k][8];
      }
    }
    __syncthreads();
    // ---- WMMA over this 128-wide K chunk: 4 steps of K=32 ----
#pragma unroll
    for (int ks = 0; ks < 4; ++ks) {
      frag16 bf;                                        // B frag: column wcol, K kb / kb+16
      const _Float16* bp = bt + btrow + (size_t)(c * 128 + ks * 32 + ((lane >> 4) << 3));
      bf.q[0] = *(const v4u*)(bp);
      bf.q[1] = *(const v4u*)(bp + 16);
#pragma unroll
      for (int k = 0; k < KP; ++k) {                    // KP independent accumulators
        frag16 af;                                      // A frag: row Mrow, K kb / kb+16
        const _Float16* ap = &As[(size_t)(k * 32 + Mrow) * AS_STRIDE + ks * 32 + kb_lane];
        af.q[0] = *(const v4u*)(ap);
        af.q[1] = *(const v4u*)(ap + 16);
        acc[k] = __builtin_amdgcn_wmma_f32_16x16x32_f16(
            false, af.h, false, bf.h, (short)0, acc[k], false, false);
      }
    }
    __syncthreads();
  }
}

template <int LO>
__device__ __forceinline__ void store_out(const v8f* acc, int zbase, float* __restrict__ out) {
  constexpr int KP = 2 * LO + 1;
  constexpr int OFFO = seg_off(LO);
  const int tid = threadIdx.x, lane = tid & 31, wave = tid >> 5;
  const int mi = wave >> 2, ni = wave & 3;
  const int n = lane & 15;
  const int mhi = (lane & 16) ? 8 : 0;                  // C/D layout: VGPR r -> M=r(+8)
  const int w = ni * 16 + n;
  const float scale = 1.0f / 24.0f;                     // 1/sqrt(576)
#pragma unroll
  for (int k = 0; k < KP; ++k)
#pragma unroll
    for (int r = 0; r < 8; ++r) {
      int z = zbase + mi * 16 + r + mhi;
      out[(size_t)z * 576 + OFFO + w * KP + k] = acc[k][r] * scale;
    }
}

__global__ __launch_bounds__(256) void tp_kernel(
    const float* __restrict__ x1, const float* __restrict__ x2,
    const float* __restrict__ cgws, const _Float16* __restrict__ bt,
    float* __restrict__ out)
{
  __shared__ __align__(16) _Float16 As[5 * 32 * AS_STRIDE];   // 43.5 KB
  __shared__ float cg_lds[CG_STRIDE];
  const int iog = blockIdx.x % 3;                       // output-l group (disjoint segments)
  const int zbase = (blockIdx.x / 3) * 32;
  v8f acc[5];
#pragma unroll
  for (int k = 0; k < 5; ++k) acc[k] = (v8f){0, 0, 0, 0, 0, 0, 0, 0};

  if (iog == 0) {            // lo=0: paths (0,0,0),(1,1,0),(2,2,0)
    process_path<0, 0, 0>(0, zbase, x1, x2, cgws, bt, As, cg_lds, acc);
    process_path<1, 1, 0>(1, zbase, x1, x2, cgws, bt, As, cg_lds, acc);
    process_path<2, 2, 0>(2, zbase, x1, x2, cgws, bt, As, cg_lds, acc);
    store_out<0>(acc, zbase, out);
  } else if (iog == 1) {     // lo=1: paths (0,1,1),(1,0,1),(1,2,1),(2,1,1)
    process_path<0, 1, 1>(3, zbase, x1, x2, cgws, bt, As, cg_lds, acc);
    process_path<1, 0, 1>(4, zbase, x1, x2, cgws, bt, As, cg_lds, acc);
    process_path<1, 2, 1>(5, zbase, x1, x2, cgws, bt, As, cg_lds, acc);
    process_path<2, 1, 1>(6, zbase, x1, x2, cgws, bt, As, cg_lds, acc);
    store_out<1>(acc, zbase, out);
  } else {                   // lo=2: paths (0,2,2),(1,1,2),(2,0,2),(2,2,2)
    process_path<0, 2, 2>(7,  zbase, x1, x2, cgws, bt, As, cg_lds, acc);
    process_path<1, 1, 2>(8,  zbase, x1, x2, cgws, bt, As, cg_lds, acc);
    process_path<2, 0, 2>(9,  zbase, x1, x2, cgws, bt, As, cg_lds, acc);
    process_path<2, 2, 2>(10, zbase, x1, x2, cgws, bt, As, cg_lds, acc);
    store_out<2>(acc, zbase, out);
  }
}

// ---------------------------------------------------------------------------
// Workspace layout: [0, 5632)           : CG tables (11 paths x 128 floats)
//                   [8192, 8192+5.5MB)  : f16 transposed weights Bt[p][w][uv]
// All launches on `stream`, deterministic, graph-capture safe.
// ---------------------------------------------------------------------------
extern "C" void kernel_launch(void* const* d_in, const int* in_sizes, int n_in,
                              void* d_out, int out_size, void* d_ws, size_t ws_size,
                              hipStream_t stream) {
  (void)in_sizes; (void)n_in; (void)out_size; (void)ws_size;
  const float* x1 = (const float*)d_in[0];
  const float* x2 = (const float*)d_in[1];
  const float* W  = (const float*)d_in[2];
  float* out = (float*)d_out;
  float*     cgws = (float*)d_ws;
  _Float16*  bt   = (_Float16*)((char*)d_ws + BT_OFFSET);

  cg_setup<<<1, 16, 0, stream>>>(cgws);
  const int nconv = 11 * 64 * 4096;
  wconv<<<(nconv + 255) / 256, 256, 0, stream>>>(W, bt);
  tp_kernel<<<384, 256, 0, stream>>>(x1, x2, cgws, bt, out);
}
